// DiagSSMBlock_30099130810508
// MI455X (gfx1250) — compile-verified, matrix operand order checked by reading, not run on hardware
//
#include <hip/hip_runtime.h>
#include <stdint.h>

#define T_SEQ 8192
#define H_DIM 1024
#define N_CHUNK 64
#define L_CHUNK 128           // N_CHUNK * L_CHUNK == T_SEQ
#define NK_ITER (H_DIM / 32)  // 32 K-steps of 32

typedef __attribute__((ext_vector_type(16))) __bf16 v16bf;
typedef __attribute__((ext_vector_type(8)))  __bf16 v8bf;
typedef __attribute__((ext_vector_type(4)))  __bf16 v4bf;
typedef __attribute__((ext_vector_type(8)))  float  v8f;
typedef __attribute__((ext_vector_type(4)))  unsigned int u32x4;
typedef __attribute__((ext_vector_type(8)))  int i32x8;
typedef __attribute__((ext_vector_type(4)))  int i32x4;
typedef int gv4i __attribute__((vector_size(16)));   // matches builtin's V4i

#if __has_builtin(__builtin_amdgcn_global_load_async_to_lds_b128)
#define HAVE_ASYNC_B128 1
#endif
#if __has_builtin(__builtin_amdgcn_tensor_load_to_lds)
#define HAVE_TDM 1
#endif

// ---- LDS geometry: rows of 32 bf16 (64B) padded to 80B (bank-conflict-free:
// dword-bank = 20*row mod 64 covers 16 distinct groups for 16 lanes) ----
#define LDS_ROWB 80
#define LDS_A_PREC (128 * LDS_ROWB)                        // 10240 B
#define LDS_B_PREC (256 * LDS_ROWB)                        // 20480 B
#define LDS_A_OFF(buf, prec) ((((buf)*2) + (prec)) * LDS_A_PREC)
#define LDS_B_OFF(buf, prec) (4 * LDS_A_PREC + (((buf)*2) + (prec)) * LDS_B_PREC)
#define LDS_TOTAL (4 * LDS_A_PREC + 4 * LDS_B_PREC)        // 122880 B

// ---------------------------------------------------------------------------
// Prep 1: split fp32 X into bf16 hi + lo  (x = hi + lo)
// ---------------------------------------------------------------------------
__global__ __launch_bounds__(256) void k_split_x(const float* __restrict__ x,
                                                 __bf16* __restrict__ xhi,
                                                 __bf16* __restrict__ xlo) {
  int idx = (blockIdx.x * 256 + threadIdx.x) * 4;
  float4 v = *reinterpret_cast<const float4*>(x + idx);
  __bf16 h0 = (__bf16)v.x, h1 = (__bf16)v.y, h2 = (__bf16)v.z, h3 = (__bf16)v.w;
  v4bf hi; hi[0] = h0; hi[1] = h1; hi[2] = h2; hi[3] = h3;
  v4bf lo;
  lo[0] = (__bf16)(v.x - (float)h0);
  lo[1] = (__bf16)(v.y - (float)h1);
  lo[2] = (__bf16)(v.z - (float)h2);
  lo[3] = (__bf16)(v.w - (float)h3);
  *reinterpret_cast<v4bf*>(xhi + idx) = hi;
  *reinterpret_cast<v4bf*>(xlo + idx) = lo;
}

// ---------------------------------------------------------------------------
// Prep 2: Bt[n][k] = B[k][n], bf16 hi/lo, LDS-tiled transpose
// ---------------------------------------------------------------------------
__global__ __launch_bounds__(256) void k_transpose_split_b(const float* __restrict__ b,
                                                           __bf16* __restrict__ bthi,
                                                           __bf16* __restrict__ btlo) {
  __shared__ float tile[32][33];
  const int tx = threadIdx.x, ty = threadIdx.y;
  const int n0 = blockIdx.x * 32, k0 = blockIdx.y * 32;
#pragma unroll
  for (int r = 0; r < 4; ++r)
    tile[ty + r * 8][tx] = b[(size_t)(k0 + ty + r * 8) * H_DIM + n0 + tx];
  __syncthreads();
#pragma unroll
  for (int r = 0; r < 4; ++r) {
    float v = tile[tx][ty + r * 8];
    __bf16 h = (__bf16)v;
    size_t o = (size_t)(n0 + ty + r * 8) * H_DIM + k0 + tx;
    bthi[o] = h;
    btlo[o] = (__bf16)(v - (float)h);
  }
}

// ---------------------------------------------------------------------------
// Data movers
// ---------------------------------------------------------------------------
__device__ __forceinline__ void copy_b128(char* lds_dst, const __bf16* src) {
#if defined(HAVE_ASYNC_B128)
  __builtin_amdgcn_global_load_async_to_lds_b128(
      (gv4i __attribute__((address_space(1)))*)(uintptr_t)src,
      (gv4i __attribute__((address_space(3)))*)(uint32_t)(uintptr_t)lds_dst,
      0, 0);
#else
  *reinterpret_cast<uint4*>(lds_dst) = *reinterpret_cast<const uint4*>(src);
#endif
}

__device__ __forceinline__ void wait_async0() {
#if defined(HAVE_ASYNC_B128)
#if __has_builtin(__builtin_amdgcn_s_wait_asynccnt)
  __builtin_amdgcn_s_wait_asynccnt(0);
#else
  asm volatile("s_wait_asynccnt 0x0" ::: "memory");
#endif
#endif
}

__device__ __forceinline__ void wait_tensor0() {
#if defined(HAVE_TDM)
#if __has_builtin(__builtin_amdgcn_s_wait_tensorcnt)
  __builtin_amdgcn_s_wait_tensorcnt(0);
#else
  asm volatile("s_wait_tensorcnt 0x0" ::: "memory");
#endif
#endif
}

#if defined(HAVE_TDM)
// TDM: 2D tile (tile_dim0=32 elems, tile_dim1=rows) from a (tensor_rows x H_DIM)
// bf16 tensor with row stride H_DIM; pad 4 DWORDs after every 16 DWORDs so
// LDS rows land on the 80-byte stride the fragment gather expects.
__device__ __forceinline__ void tdm_load_rows(const __bf16* src_tile,
                                              uint32_t lds_addr, int rows,
                                              int tensor_rows) {
  uint64_t ga = (uint64_t)(uintptr_t)src_tile;
  u32x4 g0;
  g0[0] = 1u;                                   // count=1, user mode
  g0[1] = lds_addr;                             // bits 63:32  lds_addr
  g0[2] = (uint32_t)ga;                         // bits 95:64  global_addr lo
  g0[3] = (uint32_t)((ga >> 32) & 0x01FFFFFFu)  // bits 120:96 global_addr hi
          | (2u << 30);                         // type = 2 ("image")
  i32x8 g1;
  g1[0] = (int)((1u << 16)      // data_size = 2B
              | (1u << 20)      // pad_enable
              | (3u << 22)      // pad_interval: 16 DWORDs
              | (3u << 25));    // pad_amount: 4 DWORDs
  g1[1] = (int)((H_DIM & 0xFFFF) << 16);                   // tensor_dim0 lo16
  g1[2] = (int)((H_DIM >> 16) | ((tensor_rows & 0xFFFF) << 16)); // dim0 hi | dim1 lo
  g1[3] = (int)(((unsigned)tensor_rows >> 16) | (32u << 16));    // dim1 hi | tile_dim0
  g1[4] = rows;                                            // tile_dim1 (tile_dim2=0)
  g1[5] = H_DIM;                                           // tensor_dim0_stride lo
  g1[6] = 0;
  g1[7] = 0;
  i32x4 z4 = {0, 0, 0, 0};
#if __clang_major__ >= 23
  i32x8 z8 = {0, 0, 0, 0, 0, 0, 0, 0};
  __builtin_amdgcn_tensor_load_to_lds(g0, g1, z4, z4, z8, 0);
#else
  __builtin_amdgcn_tensor_load_to_lds(g0, g1, z4, z4, 0);
#endif
}
#endif

// Stage one K-step: A via per-lane async-to-LDS, B via TDM (wave 0)
__device__ __forceinline__ void stage(char* smem, int buf,
                                      const __bf16* xhi, const __bf16* xlo,
                                      const __bf16* bthi, const __bf16* btlo,
                                      int m_blk, int n_blk, int k0,
                                      int tid, int wave) {
#pragma unroll
  for (int p = 0; p < 2; ++p) {
    const __bf16* src = p ? xlo : xhi;
    char* lds = smem + LDS_A_OFF(buf, p);
#pragma unroll
    for (int u = 0; u < 2; ++u) {               // 512 16B chunks / 256 thr
      const int c = tid + u * 256;
      const int row = c >> 2, seg = c & 3;
      copy_b128(lds + row * LDS_ROWB + seg * 16,
                src + (size_t)(m_blk + row) * H_DIM + k0 + seg * 8);
    }
  }
#if defined(HAVE_TDM)
  if (wave == 0) {
    tdm_load_rows(bthi + (size_t)n_blk * H_DIM + k0,
                  (uint32_t)(uintptr_t)(smem + LDS_B_OFF(buf, 0)), 256, H_DIM);
    tdm_load_rows(btlo + (size_t)n_blk * H_DIM + k0,
                  (uint32_t)(uintptr_t)(smem + LDS_B_OFF(buf, 1)), 256, H_DIM);
  }
#else
#pragma unroll
  for (int p = 0; p < 2; ++p) {
    const __bf16* src = p ? btlo : bthi;
    char* lds = smem + LDS_B_OFF(buf, p);
#pragma unroll
    for (int u = 0; u < 4; ++u) {               // 1024 16B chunks / 256 thr
      const int c = tid + u * 256;
      const int row = c >> 2, seg = c & 3;
      copy_b128(lds + row * LDS_ROWB + seg * 16,
                src + (size_t)(n_blk + row) * H_DIM + k0 + seg * 8);
    }
  }
#endif
}

// ---------------------------------------------------------------------------
// GEMM: S = X * B, bf16x3 split on v_wmma_f32_16x16x32_bf16, LDS-staged.
// Fragment layout (ISA 7.12.2): lane<16 -> row=lane, K {0..7,16..23};
// lane>=16 -> row=lane-16, K {8..15,24..31}.
// ---------------------------------------------------------------------------
__device__ __forceinline__ v16bf load_frag(const __bf16* p) {
  v8bf a = *reinterpret_cast<const v8bf*>(p);        // K base+0..7
  v8bf b = *reinterpret_cast<const v8bf*>(p + 16);   // K base+16..23
  v16bf r;
#pragma unroll
  for (int i = 0; i < 8; ++i) { r[i] = a[i]; r[i + 8] = b[i]; }
  return r;
}

__global__ __launch_bounds__(256) void k_gemm_bf16x3(const __bf16* __restrict__ xhi,
                                                     const __bf16* __restrict__ xlo,
                                                     const __bf16* __restrict__ bthi,
                                                     const __bf16* __restrict__ btlo,
                                                     float* __restrict__ s) {
  extern __shared__ char smem[];
  const int tid  = threadIdx.x;
  const int lane = tid & 31;
  const int wave = tid >> 5;            // 8 waves: 2 (M) x 4 (N)
  const int wm   = wave >> 2;
  const int wn   = wave & 3;
  const int m_blk = blockIdx.x * 128;
  const int n_blk = blockIdx.y * 256;
  const int l16   = lane & 15;
  const int half8 = (lane >> 4) * 8;

  v8f acc[4][4] = {};

  stage(smem, 0, xhi, xlo, bthi, btlo, m_blk, n_blk, 0, tid, wave);
  wait_async0();
  wait_tensor0();
  __syncthreads();

  for (int kk = 0; kk < NK_ITER; ++kk) {
    const int cur = kk & 1;
    if (kk + 1 < NK_ITER)
      stage(smem, cur ^ 1, xhi, xlo, bthi, btlo, m_blk, n_blk, (kk + 1) * 32,
            tid, wave);

    const char* aHiB = smem + LDS_A_OFF(cur, 0);
    const char* aLoB = smem + LDS_A_OFF(cur, 1);
    const char* bHiB = smem + LDS_B_OFF(cur, 0);
    const char* bLoB = smem + LDS_B_OFF(cur, 1);

    v16bf ahi[4], alo[4];
#pragma unroll
    for (int i = 0; i < 4; ++i) {
      const int row = wm * 64 + i * 16 + l16;
      ahi[i] = load_frag((const __bf16*)(aHiB + row * LDS_ROWB) + half8);
      alo[i] = load_frag((const __bf16*)(aLoB + row * LDS_ROWB) + half8);
    }
#pragma unroll
    for (int j = 0; j < 4; ++j) {
      const int rowb = wn * 64 + j * 16 + l16;
      v16bf bhi = load_frag((const __bf16*)(bHiB + rowb * LDS_ROWB) + half8);
      v16bf blo = load_frag((const __bf16*)(bLoB + rowb * LDS_ROWB) + half8);
#pragma unroll
      for (int i = 0; i < 4; ++i) {
        acc[i][j] = __builtin_amdgcn_wmma_f32_16x16x32_bf16(
            false, ahi[i], false, bhi, (short)0, acc[i][j], false, false);
        acc[i][j] = __builtin_amdgcn_wmma_f32_16x16x32_bf16(
            false, alo[i], false, bhi, (short)0, acc[i][j], false, false);
        acc[i][j] = __builtin_amdgcn_wmma_f32_16x16x32_bf16(
            false, ahi[i], false, blo, (short)0, acc[i][j], false, false);
      }
    }
    wait_async0();
    wait_tensor0();
    __syncthreads();
  }

  const int m_base = m_blk + wm * 64;
  const int n_base = n_blk + wn * 64;
#pragma unroll
  for (int i = 0; i < 4; ++i)
#pragma unroll
    for (int j = 0; j < 4; ++j)
#pragma unroll
      for (int r = 0; r < 8; ++r)
        s[(size_t)(m_base + i * 16 + half8 + r) * H_DIM + n_base + j * 16 + l16] =
            acc[i][j][r];
}

// ---------------------------------------------------------------------------
// Chunked parallel scan (h_t = a*h_{t-1} + s_t), 3 phases
// ---------------------------------------------------------------------------
__global__ __launch_bounds__(256) void k_scan_local(const float* __restrict__ s,
                                                    const float* __restrict__ a,
                                                    float* __restrict__ out,
                                                    float* __restrict__ f) {
  const int ch = blockIdx.y * 256 + threadIdx.x;
  const int c  = blockIdx.x;
  const float av = a[ch];
  float acc = 0.0f;
  const int t0 = c * L_CHUNK;
  for (int t = 0; t < L_CHUNK; ++t) {
    const size_t idx = (size_t)(t0 + t) * H_DIM + ch;
    acc = fmaf(av, acc, s[idx]);
    out[idx] = acc;
  }
  f[(size_t)c * H_DIM + ch] = acc;
}

__global__ __launch_bounds__(256) void k_scan_carry(const float* __restrict__ f,
                                                    const float* __restrict__ a,
                                                    float* __restrict__ P) {
  const int ch = blockIdx.x * 256 + threadIdx.x;
  const float av = a[ch];
  float aL = av;
#pragma unroll
  for (int i = 0; i < 7; ++i) aL *= aL;  // a^128
  float p = 0.0f;
  for (int c = 0; c < N_CHUNK; ++c) {
    P[(size_t)c * H_DIM + ch] = p;
    p = fmaf(aL, p, f[(size_t)c * H_DIM + ch]);
  }
}

__global__ __launch_bounds__(256) void k_scan_fix(const float* __restrict__ P,
                                                  const float* __restrict__ a,
                                                  float* __restrict__ out) {
  const int ch = blockIdx.y * 256 + threadIdx.x;
  const int c  = blockIdx.x;
  const float av = a[ch];
  const float p = P[(size_t)c * H_DIM + ch];
  float pw = av;
  const int t0 = c * L_CHUNK;
  for (int t = 0; t < L_CHUNK; ++t) {
    const size_t idx = (size_t)(t0 + t) * H_DIM + ch;
    out[idx] = fmaf(pw, p, out[idx]);
    pw *= av;
  }
}

// ---------------------------------------------------------------------------
extern "C" void kernel_launch(void* const* d_in, const int* in_sizes, int n_in,
                              void* d_out, int out_size, void* d_ws, size_t ws_size,
                              hipStream_t stream) {
  const float* x = (const float*)d_in[0];   // (T, H) fp32
  const float* a = (const float*)d_in[1];   // (H,)   fp32
  const float* b = (const float*)d_in[2];   // (H, H) fp32
  float* out = (float*)d_out;               // (T, H) fp32

  char* ws = (char*)d_ws;
  float*  S    = (float*)(ws);                                   // 32 MB
  __bf16* xhi  = (__bf16*)(ws + ((size_t)32 << 20));             // 16 MB
  __bf16* xlo  = (__bf16*)(ws + ((size_t)48 << 20));             // 16 MB
  __bf16* bthi = (__bf16*)(ws + ((size_t)64 << 20));             //  2 MB
  __bf16* btlo = (__bf16*)(ws + ((size_t)66 << 20));             //  2 MB
  float*  f    = (float*)(ws + ((size_t)68 << 20));              // 256 KB
  float*  P    = (float*)(ws + ((size_t)68 << 20) + N_CHUNK * H_DIM * 4);

  (void)hipFuncSetAttribute(reinterpret_cast<const void*>(k_gemm_bf16x3),
                            hipFuncAttributeMaxDynamicSharedMemorySize, LDS_TOTAL);

  k_split_x<<<(T_SEQ * H_DIM) / 1024, 256, 0, stream>>>(x, xhi, xlo);
  k_transpose_split_b<<<dim3(H_DIM / 32, H_DIM / 32), dim3(32, 8), 0, stream>>>(b, bthi, btlo);
  k_gemm_bf16x3<<<dim3(T_SEQ / 128, H_DIM / 256), 256, LDS_TOTAL, stream>>>(xhi, xlo, bthi, btlo, S);
  k_scan_local<<<dim3(N_CHUNK, H_DIM / 256), 256, 0, stream>>>(S, a, out, f);
  k_scan_carry<<<H_DIM / 256, 256, 0, stream>>>(f, a, P);
  k_scan_fix<<<dim3(N_CHUNK, H_DIM / 256), 256, 0, stream>>>(P, a, out);
}
